// Model_90366111907856
// MI455X (gfx1250) — compile-verified, matrix operand order checked by reading, not run on hardware
//
#include <hip/hip_runtime.h>
#include <hip/hip_bf16.h>
#include <math.h>

// Problem sizes (match reference)
#define NU   6000      // users / items (square tables)
#define DD   768       // embedding dim
#define HH   256       // hidden dim
#define BB   128       // batch
#define KK   120       // top-k neighbors
#define EPS_LN 1e-5f

#define GN   5         // n-tiles per wave in sims kernel (5*16=80 cols/block)
#define KPAD 40        // LDS row pitch in bf16 (80B: 16B-aligned, conflict-free)

typedef __attribute__((ext_vector_type(16))) __bf16 bf16x16;
typedef __attribute__((ext_vector_type(8)))  __bf16 bf16x8;
typedef __attribute__((ext_vector_type(8)))  float  f32x8;
typedef __attribute__((ext_vector_type(4)))  int    v4i;

#define AS1 __attribute__((address_space(1)))
#define AS3 __attribute__((address_space(3)))

// Async global->LDS path (gfx1250): guarded so the file compiles either way.
#if defined(__gfx1250__) && __has_builtin(__builtin_amdgcn_global_load_async_to_lds_b128) && __has_builtin(__builtin_amdgcn_s_wait_asynccnt)
#define USE_ASYNC_LDS 1
#else
#define USE_ASYNC_LDS 0
#endif

// ---------------------------------------------------------------------------
// WMMA fragment helpers.  16-bit A 16x32 layout (ISA 7.12.2): lanes 0-15 hold
// M=lane, K={0..7,16..23}; lanes 16-31 hold M=lane-16, K={8..15,24..31}.
// Per lane: kb=(lane>>4)*8, two contiguous 8-element runs at k0+kb, k0+kb+16.
// B mirrors this with N in place of M (we store B with N as the row dim).
// ---------------------------------------------------------------------------
__device__ __forceinline__ bf16x16 frag_f32_full(const float* __restrict__ rowptr,
                                                 int k0) {
    const int kb = ((threadIdx.x & 31) >> 4) * 8;
    bf16x16 f;
#pragma unroll
    for (int g = 0; g < 2; ++g) {
        const float4 a = *reinterpret_cast<const float4*>(rowptr + k0 + kb + g * 16);
        const float4 b = *reinterpret_cast<const float4*>(rowptr + k0 + kb + g * 16 + 4);
        f[g*8+0] = (__bf16)a.x; f[g*8+1] = (__bf16)a.y;
        f[g*8+2] = (__bf16)a.z; f[g*8+3] = (__bf16)a.w;
        f[g*8+4] = (__bf16)b.x; f[g*8+5] = (__bf16)b.y;
        f[g*8+6] = (__bf16)b.z; f[g*8+7] = (__bf16)b.w;
    }
    return f;
}

__device__ __forceinline__ bf16x16 frag_bf16_full(const __bf16* __restrict__ rowptr,
                                                  int k0) {
    const int kb = ((threadIdx.x & 31) >> 4) * 8;
    bf16x16 f;
#pragma unroll
    for (int g = 0; g < 2; ++g) {
        bf16x8 v = *reinterpret_cast<const bf16x8*>(rowptr + k0 + kb + g * 16);
#pragma unroll
        for (int j = 0; j < 8; ++j) f[g*8+j] = v[j];
    }
    return f;
}

__device__ __forceinline__ bf16x16 frag_bf16_guard(const __bf16* __restrict__ rowptr,
                                                   int k0, int kmax) {
    const int kb = ((threadIdx.x & 31) >> 4) * 8;
    bf16x16 f;
#pragma unroll
    for (int g = 0; g < 2; ++g) {
#pragma unroll
        for (int j = 0; j < 8; ++j) {
            const int k = k0 + kb + g * 16 + j;
            f[g*8+j] = (k < kmax) ? rowptr[k] : (__bf16)0.0f;
        }
    }
    return f;
}

// B fragment straight out of an LDS tile stored as [n(16)][KPAD] bf16.
__device__ __forceinline__ bf16x16 frag_lds(const __bf16* tile) {
    const int lane = threadIdx.x & 31;
    const int n  = lane & 15;
    const int kb = (lane >> 4) * 8;
    bf16x16 f;
#pragma unroll
    for (int g = 0; g < 2; ++g) {
        bf16x8 v = *reinterpret_cast<const bf16x8*>(tile + n * KPAD + kb + g * 16);
#pragma unroll
        for (int j = 0; j < 8; ++j) f[g*8+j] = v[j];
    }
    return f;
}

// C/D 16x16 f32 layout: VGPR r, lanes 0-15 -> M=r, lanes 16-31 -> M=r+8; N=lane&15.
#define TILE_MN()  const int c_n = (threadIdx.x & 15); const int c_mo = ((threadIdx.x & 31) >> 4) * 8

// ---------------------------------------------------------------------------
// K1: per-row inverse L2 norms (the /5 cancels in cosine similarity).
// ---------------------------------------------------------------------------
__global__ void rownorm_kernel(const float* __restrict__ ur,
                               const float* __restrict__ ir,
                               float* __restrict__ invn) {
    const int side = blockIdx.y, row = blockIdx.x, tid = threadIdx.x;
    const float* R = (side ? ir : ur) + (size_t)row * NU;
    float s = 0.0f;
    for (int n = tid; n < NU; n += 256) { float x = R[n]; s += x * x; }
    __shared__ float red[256];
    red[tid] = s; __syncthreads();
    for (int st = 128; st > 0; st >>= 1) {
        if (tid < st) red[tid] += red[tid + st];
        __syncthreads();
    }
    if (tid == 0) invn[side * NU + row] = 1.0f / fmaxf(sqrtf(red[0]), 1e-12f);
}

// ---------------------------------------------------------------------------
// K2: pre-gather + normalize + bf16-convert the batch rows (A of sims GEMM).
// ---------------------------------------------------------------------------
__global__ void gather_a_kernel(const float* __restrict__ ur,
                                const float* __restrict__ ir,
                                const int* __restrict__ uids,
                                const int* __restrict__ iids,
                                const float* __restrict__ invn,
                                __bf16* __restrict__ Abuf) {
    const int side = blockIdx.y, b = blockIdx.x;
    const float* R   = side ? ir : ur;
    const int*   ids = side ? iids : uids;
    const int   row = ids[b];
    const float sc  = invn[side * NU + row];
    const float* src = R + (size_t)row * NU;
    __bf16* dst = Abuf + ((size_t)side * BB + b) * NU;
    for (int j = threadIdx.x; j < NU / 8; j += 256) {
        const int k = j * 8;
        float4 x = *reinterpret_cast<const float4*>(src + k);
        float4 y = *reinterpret_cast<const float4*>(src + k + 4);
        bf16x8 v;
        v[0] = (__bf16)(x.x*sc); v[1] = (__bf16)(x.y*sc);
        v[2] = (__bf16)(x.z*sc); v[3] = (__bf16)(x.w*sc);
        v[4] = (__bf16)(y.x*sc); v[5] = (__bf16)(y.y*sc);
        v[6] = (__bf16)(y.z*sc); v[7] = (__bf16)(y.w*sc);
        *reinterpret_cast<bf16x8*>(dst + k) = v;
    }
}

// ---------------------------------------------------------------------------
// K3: sims = A_bf16 @ (R*inv)^T.  grid=(75, 2), block=256 (8 waves = 8 M-tiles
// covering all 128 batch rows).  The block stages an 80x32 B chunk in LDS per
// K-step; staging is straight-line (job tid + optional job 256+tid) so there
// is no divergent loop in the steady state.  5 WMMAs per A fragment.
// ---------------------------------------------------------------------------
__device__ __forceinline__ void stage_b(const float* __restrict__ R,
                                        const float* __restrict__ inv,
                                        int n0, int k0, int j,
                                        __bf16 (*Bs)[KPAD]) {
    const int row = j >> 2, kg = (j & 3) * 8;
    const int nrow = n0 + row;
    const float sc = inv[nrow];
    const float* p = R + (size_t)nrow * NU + k0 + kg;
    __builtin_prefetch(p + 32, 0, 3);
    float4 x = *reinterpret_cast<const float4*>(p);
    float4 y = *reinterpret_cast<const float4*>(p + 4);
    __bf16* d = &Bs[row][kg];
    d[0] = (__bf16)(x.x*sc); d[1] = (__bf16)(x.y*sc);
    d[2] = (__bf16)(x.z*sc); d[3] = (__bf16)(x.w*sc);
    d[4] = (__bf16)(y.x*sc); d[5] = (__bf16)(y.y*sc);
    d[6] = (__bf16)(y.z*sc); d[7] = (__bf16)(y.w*sc);
}

__device__ __forceinline__ void stage_b_tail(const float* __restrict__ R,
                                             const float* __restrict__ inv,
                                             int n0, int k0, int j,
                                             __bf16 (*Bs)[KPAD]) {
    const int row = j >> 2, kg = (j & 3) * 8;
    __bf16* d = &Bs[row][kg];
    if (k0 + kg + 8 <= NU) {
        const int nrow = n0 + row;
        const float sc = inv[nrow];
        const float* p = R + (size_t)nrow * NU + k0 + kg;
        float4 x = *reinterpret_cast<const float4*>(p);
        float4 y = *reinterpret_cast<const float4*>(p + 4);
        d[0] = (__bf16)(x.x*sc); d[1] = (__bf16)(x.y*sc);
        d[2] = (__bf16)(x.z*sc); d[3] = (__bf16)(x.w*sc);
        d[4] = (__bf16)(y.x*sc); d[5] = (__bf16)(y.y*sc);
        d[6] = (__bf16)(y.z*sc); d[7] = (__bf16)(y.w*sc);
    } else {
#pragma unroll
        for (int q = 0; q < 8; ++q) d[q] = (__bf16)0.0f;
    }
}

__global__ void sims_wmma_kernel(const __bf16* __restrict__ Abuf,
                                 const float* __restrict__ ur,
                                 const float* __restrict__ ir,
                                 const float* __restrict__ invn,
                                 float* __restrict__ sims) {
    const int side = blockIdx.y;
    const float* R   = side ? ir : ur;
    const float* inv = invn + side * NU;
    const __bf16* A  = Abuf + (size_t)side * BB * NU;
    float*       S   = sims + (size_t)side * BB * NU;

    const int tid  = threadIdx.x;
    const int wave = tid >> 5;
    const int lane = tid & 31;
    const int n0 = blockIdx.x * (16 * GN);
    const int m0 = wave * 16;

    __shared__ __align__(16) __bf16 Bs[16 * GN][KPAD];

    const __bf16* arow = A + (size_t)(m0 + (lane & 15)) * NU;

    f32x8 c[GN];
#pragma unroll
    for (int g = 0; g < GN; ++g) c[g] = (f32x8){};

    const int KMAIN = (NU / 32) * 32;           // 5984
    for (int k0 = 0; k0 < KMAIN; k0 += 32) {
        stage_b(R, inv, n0, k0, tid, Bs);                    // 256 of 320 jobs
        if (tid < 16 * GN * 4 - 256) stage_b(R, inv, n0, k0, 256 + tid, Bs);
        __syncthreads();
        const bf16x16 a = frag_bf16_full(arow, k0);
#pragma unroll
        for (int g = 0; g < GN; ++g) {
            const bf16x16 b = frag_lds(&Bs[g * 16][0]);
            c[g] = __builtin_amdgcn_wmma_f32_16x16x32_bf16(false, a, false, b,
                                                           (short)0, c[g], false, false);
        }
        __syncthreads();
    }
    {   // K-tail: k0 = 5984, 16 valid columns
        const int k0 = KMAIN;
        stage_b_tail(R, inv, n0, k0, tid, Bs);
        if (tid < 16 * GN * 4 - 256) stage_b_tail(R, inv, n0, k0, 256 + tid, Bs);
        __syncthreads();
        const bf16x16 a = frag_bf16_guard(arow, k0, NU);
#pragma unroll
        for (int g = 0; g < GN; ++g) {
            const bf16x16 b = frag_lds(&Bs[g * 16][0]);
            c[g] = __builtin_amdgcn_wmma_f32_16x16x32_bf16(false, a, false, b,
                                                           (short)0, c[g], false, false);
        }
    }
    TILE_MN();
#pragma unroll
    for (int g = 0; g < GN; ++g)
#pragma unroll
        for (int r = 0; r < 8; ++r)
            S[(size_t)(m0 + c_mo + r) * NU + (n0 + g * 16 + c_n)] = c[g][r];
}

// ---------------------------------------------------------------------------
// K4: zero the union-of-neighbors masks.
// ---------------------------------------------------------------------------
__global__ void zero_uniq_kernel(int* __restrict__ uniq) {
    int i = blockIdx.x * 256 + threadIdx.x;
    if (i < 2 * NU) uniq[i] = 0;
}

// ---------------------------------------------------------------------------
// K5: Gumbel top-K + masked softmax; one block per (side,row); keys in LDS.
// ---------------------------------------------------------------------------
__device__ __forceinline__ unsigned mix32(unsigned x) {
    x ^= x >> 16; x *= 0x7feb352du;
    x ^= x >> 15; x *= 0x846ca68bu;
    x ^= x >> 16; return x;
}

__global__ void topk_softmax_kernel(const float* __restrict__ sims,
                                    const int* __restrict__ uids,
                                    const int* __restrict__ iids,
                                    float* __restrict__ probs,
                                    int* __restrict__ uniq) {
    const int side = blockIdx.y, b = blockIdx.x, tid = threadIdx.x;
    const float* S = sims  + ((size_t)side * BB + b) * NU;
    float*       P = probs + ((size_t)side * BB + b) * NU;
    int*        Uq = uniq + side * NU;

    __shared__ float    keys[NU];
    __shared__ unsigned selbits[(NU + 31) / 32];
    __shared__ float    rv[256];
    __shared__ int      ri[256];

    for (int n = tid; n < NU; n += 256) {
        unsigned h = mix32((unsigned)(b * NU + n) * 0x9e3779b9u ^ (side ? 0x85ebca6bu : 0x27d4eb2fu));
        float u = ((h >> 8) + 1u) * (1.0f / 16777216.0f);        // (0,1]
        keys[n] = S[n] + (-logf(-logf(u)));                      // + Gumbel
    }
    for (int w = tid; w < (NU + 31) / 32; w += 256) selbits[w] = 0u;
    __syncthreads();

    for (int it = 0; it < KK; ++it) {
        float best = -__builtin_inff(); int bi = 0;
        for (int n = tid; n < NU; n += 256)
            if (keys[n] > best) { best = keys[n]; bi = n; }
        rv[tid] = best; ri[tid] = bi;
        __syncthreads();
        for (int st = 128; st > 0; st >>= 1) {
            if (tid < st && rv[tid + st] > rv[tid]) { rv[tid] = rv[tid + st]; ri[tid] = ri[tid + st]; }
            __syncthreads();
        }
        if (tid == 0) {
            int a = ri[0];
            selbits[a >> 5] |= (1u << (a & 31));
            keys[a] = -__builtin_inff();
        }
        __syncthreads();
    }

    for (int n = tid; n < NU; n += 256)
        if ((selbits[n >> 5] >> (n & 31)) & 1u) atomicOr(&Uq[n], 1);
    if (tid == 0) atomicOr(&Uq[side ? iids[b] : uids[b]], 1);

    // softmax(where(sel, sims, 0)) -- unselected contribute exp(0)
    float mx = 0.0f;
    for (int n = tid; n < NU; n += 256) {
        float v = ((selbits[n >> 5] >> (n & 31)) & 1u) ? S[n] : 0.0f;
        mx = fmaxf(mx, v);
    }
    rv[tid] = mx; __syncthreads();
    for (int st = 128; st > 0; st >>= 1) {
        if (tid < st) rv[tid] = fmaxf(rv[tid], rv[tid + st]);
        __syncthreads();
    }
    mx = rv[0]; __syncthreads();

    float sum = 0.0f;
    for (int n = tid; n < NU; n += 256) {
        float v = ((selbits[n >> 5] >> (n & 31)) & 1u) ? S[n] : 0.0f;
        sum += expf(v - mx);
    }
    rv[tid] = sum; __syncthreads();
    for (int st = 128; st > 0; st >>= 1) {
        if (tid < st) rv[tid] += rv[tid + st];
        __syncthreads();
    }
    const float invs = 1.0f / rv[0];
    for (int n = tid; n < NU; n += 256) {
        float v = ((selbits[n >> 5] >> (n & 31)) & 1u) ? S[n] : 0.0f;
        P[n] = expf(v - mx) * invs;
    }
}

// ---------------------------------------------------------------------------
// K6: apply cross-row union mask, emit bf16 probs (A operand of agg GEMM).
// ---------------------------------------------------------------------------
__global__ void probs_mask_kernel(const float* __restrict__ probs,
                                  const int* __restrict__ uniq,
                                  __bf16* __restrict__ probs_bf) {
    size_t i = (size_t)blockIdx.x * 256 + threadIdx.x;   // exactly 2*BB*NU
    int n    = (int)(i % NU);
    int side = (int)(i / ((size_t)BB * NU));
    float p = probs[i] * (uniq[side * NU + n] ? 1.0f : 0.0f);
    probs_bf[i] = (__bf16)p;
}

// ---------------------------------------------------------------------------
// K7: one-shot f32->bf16 conversion of the 4 live fcW tables (1.5 MB; reused
// by 375 M-blocks, so converting once beats in-loop conversion 375x).
// ---------------------------------------------------------------------------
__global__ void wconv_kernel(const float* __restrict__ fcW,
                             __bf16* __restrict__ Wbf) {
    const int j   = blockIdx.x * 256 + threadIdx.x;     // 4*HH*DD/8 jobs
    const int per = (HH * DD) / 8;
    const int slot = j / per;
    const int w    = j % per;
    const int t    = (slot < 3) ? slot : 4;
    const float* src = fcW + (size_t)t * HH * DD + (size_t)w * 8;
    __bf16*      dst = Wbf + (size_t)slot * HH * DD + (size_t)w * 8;
    float4 x = *reinterpret_cast<const float4*>(src);
    float4 y = *reinterpret_cast<const float4*>(src + 4);
    bf16x8 v;
    v[0] = (__bf16)x.x; v[1] = (__bf16)x.y; v[2] = (__bf16)x.z; v[3] = (__bf16)x.w;
    v[4] = (__bf16)y.x; v[5] = (__bf16)y.y; v[6] = (__bf16)y.z; v[7] = (__bf16)y.w;
    *reinterpret_cast<bf16x8*>(dst) = v;
}

// ---------------------------------------------------------------------------
// K8: FC GEMM  h = emb[t] @ fcW[t]^T + fcb[t]  (tables 0,1,2,4 only).
// grid=(375, 4), block=128 (4 waves); each wave owns 4 N-tile accumulators.
// A converts f32->bf16 in flight (streamed once); B is pre-converted bf16.
// K=768 exactly -> no tail, no guards.
// ---------------------------------------------------------------------------
__global__ void fc_gemm_kernel(const float* __restrict__ emb,
                               const __bf16* __restrict__ Wbf,
                               const float* __restrict__ fcb,
                               float* __restrict__ hbuf) {
    const int slot = blockIdx.y;
    const int t    = (slot < 3) ? slot : 4;
    const float*  A  = emb + (size_t)t * NU * DD;
    const __bf16* W  = Wbf + (size_t)slot * HH * DD;
    const float*  bs = fcb + t * HH;
    float* Hout = hbuf + (size_t)slot * NU * HH;

    const int wave = threadIdx.x >> 5;
    const int lane = threadIdx.x & 31;
    const int m0 = blockIdx.x * 16;
    const float* arow = A + (size_t)(m0 + (lane & 15)) * DD;
    const __bf16* brow[4];
#pragma unroll
    for (int g = 0; g < 4; ++g)
        brow[g] = W + (size_t)((wave * 4 + g) * 16 + (lane & 15)) * DD;   // B(k,n)=fcW[n][k]

    f32x8 c[4];
#pragma unroll
    for (int g = 0; g < 4; ++g) c[g] = (f32x8){};

    for (int k0 = 0; k0 < DD; k0 += 32) {
        __builtin_prefetch(arow + k0 + 64, 0, 3);
        const bf16x16 a = frag_f32_full(arow, k0);
#pragma unroll
        for (int g = 0; g < 4; ++g) {
            const bf16x16 b = frag_bf16_full(brow[g], k0);
            c[g] = __builtin_amdgcn_wmma_f32_16x16x32_bf16(false, a, false, b,
                                                           (short)0, c[g], false, false);
        }
    }
    TILE_MN();
#pragma unroll
    for (int g = 0; g < 4; ++g) {
        const int n = (wave * 4 + g) * 16 + c_n;
        const float bn = bs[n];
#pragma unroll
        for (int r = 0; r < 8; ++r)
            Hout[(size_t)(m0 + c_mo + r) * HH + n] = c[g][r] + bn;
    }
}

// ---------------------------------------------------------------------------
// K9: LayerNorm + SiLU; emit feats row-major f32 (gather path) and transposed
// bf16 (K-contiguous B operand of the agg GEMM).
// ---------------------------------------------------------------------------
__global__ void ln_silu_kernel(const float* __restrict__ hbuf,
                               const float* __restrict__ fc_g,
                               const float* __restrict__ fc_beta,
                               float* __restrict__ feats,
                               __bf16* __restrict__ featsT) {
    const int slot = blockIdx.y;
    const int t    = (slot < 3) ? slot : 4;
    const int row  = blockIdx.x;
    const int h    = threadIdx.x;     // 256 == HH
    const float v = hbuf[((size_t)slot * NU + row) * HH + h];

    __shared__ float red[256];
    red[h] = v; __syncthreads();
    for (int st = 128; st > 0; st >>= 1) {
        if (h < st) red[h] += red[h + st];
        __syncthreads();
    }
    const float mu = red[0] * (1.0f / HH);
    __syncthreads();
    const float d = v - mu;
    red[h] = d * d; __syncthreads();
    for (int st = 128; st > 0; st >>= 1) {
        if (h < st) red[h] += red[h + st];
        __syncthreads();
    }
    const float var = red[0] * (1.0f / HH);
    float y = d * rsqrtf(var + EPS_LN) * fc_g[t * HH + h] + fc_beta[t * HH + h];
    float s = y / (1.0f + expf(-y));                 // SiLU
    feats[((size_t)slot * NU + row) * HH + h] = s;
    featsT[((size_t)slot * HH + h) * NU + row] = (__bf16)s;
}

// ---------------------------------------------------------------------------
// K10: agg[slot] = probs(side(slot)) @ feats[slot]  (all-bf16 operands).
// grid=(4 ngroups, 4 slots), block=256 (8 waves = 8 M-tiles); 4 N-tile accs.
// B chunk (64x32 bf16 = 4KB) staged in LDS: exactly one 16B copy per thread
// per K-step, via global_load_async_to_lds_b128 when available (ASYNCcnt
// path), else a register bounce.  Removes the 8x redundant global B reads.
// ---------------------------------------------------------------------------
__global__ void agg_gemm_kernel(const __bf16* __restrict__ probs_bf,
                                const __bf16* __restrict__ featsT,
                                float* __restrict__ agg) {
    const int slot = blockIdx.y;
    const int side = (slot == 1) ? 1 : 0;
    const __bf16* Pb = probs_bf + (size_t)side * BB * NU;
    const __bf16* Bt = featsT  + (size_t)slot * HH * NU;
    float* Aout = agg + (size_t)slot * BB * HH;

    const int tid  = threadIdx.x;
    const int wave = tid >> 5;
    const int lane = tid & 31;
    const int m0 = wave * 16;
    const int ng = blockIdx.x;

    __shared__ __align__(16) __bf16 Bs[4][16][KPAD];

    // Staging job: thread -> (tile, row, 16B k-group); 256 jobs exactly.
    const int s_tile = tid >> 6;          // 0..3
    const int s_row  = (tid >> 2) & 15;   // 0..15
    const int s_kg   = tid & 3;           // 0..3
    const __bf16* s_src = Bt + (size_t)((ng * 4 + s_tile) * 16 + s_row) * NU + s_kg * 8;
    __bf16* s_dst = &Bs[s_tile][s_row][s_kg * 8];

    const __bf16* arow = Pb + (size_t)(m0 + (lane & 15)) * NU;

    f32x8 c[4];
#pragma unroll
    for (int g = 0; g < 4; ++g) c[g] = (f32x8){};

    const int KMAIN = (NU / 32) * 32;
    for (int k0 = 0; k0 < KMAIN; k0 += 32) {
#if USE_ASYNC_LDS
        __builtin_amdgcn_global_load_async_to_lds_b128(
            (AS1 v4i*)(s_src + k0),
            (AS3 v4i*)s_dst, 0, 0);
        __builtin_amdgcn_s_wait_asynccnt(0);
#else
        *reinterpret_cast<bf16x8*>(s_dst) =
            *reinterpret_cast<const bf16x8*>(s_src + k0);
#endif
        __syncthreads();
        const bf16x16 a = frag_bf16_full(arow, k0);
#pragma unroll
        for (int g = 0; g < 4; ++g) {
            const bf16x16 b = frag_lds(&Bs[g][0][0]);
            c[g] = __builtin_amdgcn_wmma_f32_16x16x32_bf16(false, a, false, b,
                                                           (short)0, c[g], false, false);
        }
        __syncthreads();
    }
    {   // K-tail (16 valid): direct guarded global frags, no LDS
        const bf16x16 a = frag_bf16_guard(arow, KMAIN, NU);
#pragma unroll
        for (int g = 0; g < 4; ++g) {
            const __bf16* brow = Bt + (size_t)((ng * 4 + g) * 16 + (lane & 15)) * NU;
            const bf16x16 b = frag_bf16_guard(brow, KMAIN, NU);
            c[g] = __builtin_amdgcn_wmma_f32_16x16x32_bf16(false, a, false, b,
                                                           (short)0, c[g], false, false);
        }
    }
    TILE_MN();
#pragma unroll
    for (int g = 0; g < 4; ++g) {
        const int n = (ng * 4 + g) * 16 + c_n;
#pragma unroll
        for (int r = 0; r < 8; ++r)
            Aout[(size_t)(m0 + c_mo + r) * HH + n] = c[g][r];
    }
}

// ---------------------------------------------------------------------------
// K11: r[slot] = agg[slot] @ gnnW[t]^T + gnnb[t] + feats[slot][ids]
// ---------------------------------------------------------------------------
__global__ void gnn_linear_kernel(const float* __restrict__ agg,
                                  const float* __restrict__ gnnW,
                                  const float* __restrict__ gnnb,
                                  const float* __restrict__ feats,
                                  const int* __restrict__ uids,
                                  const int* __restrict__ iids,
                                  float* __restrict__ rout) {
    const int gid  = blockIdx.x * 256 + threadIdx.x;   // 4*128*256
    const int h    = gid & 255;
    const int b    = (gid >> 8) & 127;
    const int slot = gid >> 15;
    const int t    = (slot < 3) ? slot : 4;
    const float4* a4 = (const float4*)(agg  + ((size_t)slot * BB + b) * HH);
    const float4* w4 = (const float4*)(gnnW + ((size_t)t * HH + h) * HH);
    float acc = gnnb[t * HH + h];
    for (int k = 0; k < HH / 4; ++k) {
        float4 av = a4[k], wv = w4[k];
        acc += av.x * wv.x + av.y * wv.y + av.z * wv.z + av.w * wv.w;
    }
    const int id = (slot == 1) ? iids[b] : uids[b];
    acc += feats[((size_t)slot * NU + id) * HH + h];
    rout[gid] = acc;
}

// ---------------------------------------------------------------------------
// K12: pred[b] = sum_h (r0 + r2 - r3)*r1 + biases, clipped to [1,5].
// ---------------------------------------------------------------------------
__global__ void final_kernel(const float* __restrict__ rout,
                             const int* __restrict__ uids,
                             const int* __restrict__ iids,
                             const float* __restrict__ ubias,
                             const float* __restrict__ ibias,
                             const int* __restrict__ clip,
                             float* __restrict__ out) {
    const int b = blockIdx.x, h = threadIdx.x;
    const float r0 = rout[((size_t)0 * BB + b) * HH + h];
    const float r1 = rout[((size_t)1 * BB + b) * HH + h];
    const float r2 = rout[((size_t)2 * BB + b) * HH + h];
    const float r3 = rout[((size_t)3 * BB + b) * HH + h];
    __shared__ float red[256];
    red[h] = (r0 + r2 - r3) * r1; __syncthreads();
    for (int st = 128; st > 0; st >>= 1) {
        if (h < st) red[h] += red[h + st];
        __syncthreads();
    }
    if (h == 0) {
        float p = red[0] + ubias[uids[b]] + ibias[iids[b]];
        if (clip[0] == 1) p = fminf(fmaxf(p, 1.0f), 5.0f);
        out[b] = p;
    }
}

// ---------------------------------------------------------------------------
// Host: workspace layout + launch sequence (all on `stream`).
// ---------------------------------------------------------------------------
static inline size_t alignup(size_t x) { return (x + 255) & ~(size_t)255; }

extern "C" void kernel_launch(void* const* d_in, const int* in_sizes, int n_in,
                              void* d_out, int out_size, void* d_ws, size_t ws_size,
                              hipStream_t stream) {
    const int*   uids  = (const int*)  d_in[0];
    const int*   iids  = (const int*)  d_in[1];
    /* d_in[2] rating: unused by forward math */
    const int*   clip  = (const int*)  d_in[3];
    const float* ur    = (const float*)d_in[4];
    const float* ir    = (const float*)d_in[5];
    const float* emb   = (const float*)d_in[6];
    const float* fcW   = (const float*)d_in[7];
    const float* fcb   = (const float*)d_in[8];
    const float* fcg   = (const float*)d_in[9];
    const float* fcbt  = (const float*)d_in[10];
    const float* gnnW  = (const float*)d_in[11];
    const float* gnnb  = (const float*)d_in[12];
    const float* ubias = (const float*)d_in[13];
    const float* ibias = (const float*)d_in[14];
    float* out = (float*)d_out;

    char* w = (char*)d_ws;
    size_t off = 0;
    float*  invn     = (float*)(w + off);  off += alignup((size_t)2 * NU * 4);
    __bf16* Abuf     = (__bf16*)(w + off); off += alignup((size_t)2 * BB * NU * 2);
    float*  sims     = (float*)(w + off);  off += alignup((size_t)2 * BB * NU * 4);
    float*  probs    = (float*)(w + off);  off += alignup((size_t)2 * BB * NU * 4);
    __bf16* probs_bf = (__bf16*)(w + off); off += alignup((size_t)2 * BB * NU * 2);
    int*    uniq     = (int*)  (w + off);  off += alignup((size_t)2 * NU * 4);
    float*  hbuf     = (float*)(w + off);  off += alignup((size_t)4 * NU * HH * 4);
    float*  feats    = (float*)(w + off);  off += alignup((size_t)4 * NU * HH * 4);
    __bf16* featsT   = (__bf16*)(w + off); off += alignup((size_t)4 * HH * NU * 2);
    __bf16* Wbf      = (__bf16*)(w + off); off += alignup((size_t)4 * HH * DD * 2);
    float*  agg      = (float*)(w + off);  off += alignup((size_t)4 * BB * HH * 4);
    float*  rout     = (float*)(w + off);  off += alignup((size_t)4 * BB * HH * 4);
    (void)ws_size; (void)n_in; (void)in_sizes; (void)out_size;

    // 1) row inverse-norms (warms L2 with the ratings tables)
    rownorm_kernel<<<dim3(NU, 2), 256, 0, stream>>>(ur, ir, invn);
    // 2) pre-gather batch rows as scaled bf16 (A operand)
    gather_a_kernel<<<dim3(BB, 2), 256, 0, stream>>>(ur, ir, uids, iids, invn, Abuf);
    // 3) cosine-similarity GEMMs: LDS-staged B, 5 WMMAs per A fragment
    sims_wmma_kernel<<<dim3(NU / (16 * GN), 2), 256, 0, stream>>>(Abuf, ur, ir, invn, sims);
    // 4-5) Gumbel top-K + masked softmax
    zero_uniq_kernel<<<(2 * NU + 255) / 256, 256, 0, stream>>>(uniq);
    topk_softmax_kernel<<<dim3(BB, 2), 256, 0, stream>>>(sims, uids, iids, probs, uniq);
    // 6) union-mask + bf16 conversion (stream order = global barrier after 5)
    probs_mask_kernel<<<(2 * BB * NU) / 256, 256, 0, stream>>>(probs, uniq, probs_bf);
    // 7) one-shot fcW -> bf16
    wconv_kernel<<<(4 * HH * DD / 8) / 256, 256, 0, stream>>>(fcW, Wbf);
    // 8) FC GEMMs (tables 0,1,2,4), 4 WMMAs per A fragment
    fc_gemm_kernel<<<dim3(NU / 16, 4), 128, 0, stream>>>(emb, Wbf, fcb, hbuf);
    // 9) LayerNorm + SiLU, emit f32 + transposed bf16
    ln_silu_kernel<<<dim3(NU, 4), 256, 0, stream>>>(hbuf, fcg, fcbt, feats, featsT);
    // 10) neighbor aggregation GEMMs (async LDS staging when available)
    agg_gemm_kernel<<<dim3(HH / 64, 4), 256, 0, stream>>>(probs_bf, featsT, agg);
    // 11) GNN linear + skip connection
    gnn_linear_kernel<<<(4 * BB * HH) / 256, 256, 0, stream>>>(agg, gnnW, gnnb, feats, uids, iids, rout);
    // 12) prediction dot product + bias + clip
    final_kernel<<<BB, 256, 0, stream>>>(rout, uids, iids, ubias, ibias, clip, out);
}
